// Attention_39049842655427
// MI455X (gfx1250) — compile-verified, matrix operand order checked by reading, not run on hardware
//
#include <hip/hip_runtime.h>

// ---------------------------------------------------------------------------
// Problem constants (B, S, D, H) = (2, 2048, 2048, 16), DK = 128
// ---------------------------------------------------------------------------
#define B_   2
#define S_   2048
#define D_   2048
#define H_   16
#define DK_  128
#define MM_  4096          // B_*S_  (GEMM M)
#define KF_  64            // D_/32  fragments along K in the dense GEMMs

typedef __bf16 bfx16 __attribute__((ext_vector_type(16)));
typedef __bf16 bfx8  __attribute__((ext_vector_type(8)));
typedef float  fx8   __attribute__((ext_vector_type(8)));

__device__ __forceinline__ fx8 fzero8() {
    fx8 z = {0.f, 0.f, 0.f, 0.f, 0.f, 0.f, 0.f, 0.f};
    return z;
}

__device__ __forceinline__ fx8 wmma_bf16(bfx16 a, bfx16 b, fx8 c) {
    // D = A(16x32 bf16) * B(32x16 bf16) + C(16x16 f32)
    return __builtin_amdgcn_wmma_f32_16x16x32_bf16(
        /*neg_a=*/false, a, /*neg_b=*/false, b,
        /*c_mod=*/(short)0, c, /*reuse_a=*/false, /*reuse_b=*/false);
}

// ISA 7.12.2 16-bit A/B fragment element mapping:
//   lane = rc + 16*hh holds e[0..7] -> k = hh*8+e ; e[8..15] -> k = 16+hh*8+(e-8)
__device__ __forceinline__ void frag_place(int rc, int kl, int& lane, int& e) {
    int hh = (kl & 15) >> 3;
    lane = rc + (hh << 4);
    e = (kl & 7) + ((kl >> 4) << 3);
}
__device__ __forceinline__ int frag_k(int hf, int e) {
    return (e < 8) ? (hf * 8 + e) : (16 + hf * 8 + (e - 8));
}

// ---------------------------------------------------------------------------
// Pre-swizzle: row-major f32 [R][K] -> bf16 WMMA fragment layout.
// Fragment enumeration: frag = (r/16)*(K/32) + (k/32); 512 bf16 per fragment,
// within-frag address = lane*16 + e.  Works for A-operands (r = matrix row)
// and for B-operands of W^T (r = output column n, value = W[n][k]).
// ---------------------------------------------------------------------------
__global__ __launch_bounds__(256) void prep_swizzle_kernel(
    const float* __restrict__ src, __bf16* __restrict__ dst, int R, int K)
{
    size_t tid = (size_t)blockIdx.x * blockDim.x + threadIdx.x;
    if (tid >= (size_t)R * K) return;
    int e    = (int)(tid & 15);
    int lane = (int)((tid >> 4) & 31);
    size_t f = tid >> 9;
    int kfr  = K >> 5;
    int fm   = (int)(f / kfr);
    int fk   = (int)(f % kfr);
    int hf   = lane >> 4;
    int row  = (fm << 4) + (lane & 15);
    int kk   = (fk << 5) + frag_k(hf, e);
    dst[tid] = (__bf16)src[(size_t)row * K + kk];
}

// ---------------------------------------------------------------------------
// Dense GEMM (fixed 4096 x 2048 x 2048):
//   out = Af(frag layout) * Bf(frag layout of W^T) + bias
// 8 waves/block; wave tile 64(M) x 64(N) -> 16 accumulators, 16 WMMA/k-step.
// Block tile 256 x 128.  All fragment loads are base+immediate (strides are
// compile-time: 64 KiB between row tiles, 1 KiB per k-step).
// ---------------------------------------------------------------------------
__global__ __launch_bounds__(256) void gemm_bf16_kernel(
    const __bf16* __restrict__ Af, const __bf16* __restrict__ Bf,
    const float* __restrict__ bias, void* __restrict__ outp, int out_f32)
{
    const int lane = threadIdx.x & 31;
    const int wave = threadIdx.x >> 5;
    const int hf   = lane >> 4;
    const int nn   = lane & 15;
    const int NB   = D_ >> 7;            // 16 column blocks of 128
    const int bm   = blockIdx.x / NB;
    const int bn   = blockIdx.x % NB;
    const int wm   = wave >> 1;          // 0..3
    const int wn   = wave & 1;           // 0..1
    const int m0   = bm * 256 + wm * 64;
    const int n0   = bn * 128 + wn * 64;

    fx8 acc[4][4];
#pragma unroll
    for (int i = 0; i < 4; ++i)
#pragma unroll
        for (int j = 0; j < 4; ++j) acc[i][j] = fzero8();

    // byte pointers to fragment (row-tile, k=0) for this lane
    const char* pA = (const char*)(Af + ((size_t)(m0 >> 4) * KF_) * 512) + lane * 32;
    const char* pB = (const char*)(Bf + ((size_t)(n0 >> 4) * KF_) * 512) + lane * 32;
    const int ROWSTRIDE = KF_ * 1024;    // 65536 B between row tiles (imm ok)

#pragma unroll 2
    for (int fk = 0; fk < KF_; ++fk) {
        bfx16 a[4], b[4];
#pragma unroll
        for (int i = 0; i < 4; ++i) {
            a[i] = *(const bfx16*)(pA + i * ROWSTRIDE);
            b[i] = *(const bfx16*)(pB + i * ROWSTRIDE);
        }
        pA += 1024;
        pB += 1024;
#pragma unroll
        for (int i = 0; i < 4; ++i)
#pragma unroll
            for (int j = 0; j < 4; ++j)
                acc[i][j] = wmma_bf16(a[i], b[j], acc[i][j]);
    }

#pragma unroll
    for (int i = 0; i < 4; ++i)
#pragma unroll
        for (int j = 0; j < 4; ++j) {
            int ncol = n0 + j * 16 + nn;
            float bv = bias[ncol];
#pragma unroll
            for (int r = 0; r < 8; ++r) {
                int mrow = m0 + i * 16 + r + 8 * hf;   // D-layout: lane=N, vgpr=M
                float v = acc[i][j][r] + bv;
                if (out_f32) ((float*)outp)[(size_t)mrow * D_ + ncol] = v;
                else ((__bf16*)outp)[(size_t)mrow * D_ + ncol] = (__bf16)v;
            }
        }
}

// ---------------------------------------------------------------------------
// RoPE + scatter into per-head fragment layouts:
//   Qf : A-layout, per head frag = (s/16)*4 + (dk/32)       (512 frags/head)
//   Kf : B-layout of K^T, same enumeration (col = key row)
//   Vf : B-layout of V,  frag = (s/32)*8 + (dk/16)
// ---------------------------------------------------------------------------
__global__ __launch_bounds__(256) void rope_scatter_kernel(
    const __bf16* __restrict__ Qp, const __bf16* __restrict__ Kp,
    const __bf16* __restrict__ Vp, const float* __restrict__ freqs,
    __bf16* __restrict__ Qf, __bf16* __restrict__ Kf, __bf16* __restrict__ Vf)
{
    int tid = blockIdx.x * blockDim.x + threadIdx.x;   // B*H*S*(DK/2) threads
    if (tid >= B_ * H_ * S_ * (DK_ / 2)) return;
    int i = tid & 63;
    int s = (tid >> 6) & (S_ - 1);
    int h = (tid >> 17) & (H_ - 1);
    int b = tid >> 21;
    int d0 = 2 * i;

    size_t rowbase = ((size_t)(b * S_ + s)) * D_ + (size_t)h * DK_;
    float q0 = (float)Qp[rowbase + d0], q1 = (float)Qp[rowbase + d0 + 1];
    float k0 = (float)Kp[rowbase + d0], k1 = (float)Kp[rowbase + d0 + 1];
    float v0 = (float)Vp[rowbase + d0], v1 = (float)Vp[rowbase + d0 + 1];
    float ang = freqs[s * 64 + i];
    float sn, cs;
    __sincosf(ang, &sn, &cs);
    float qr0 = q0 * cs - q1 * sn, qr1 = q0 * sn + q1 * cs;
    float kr0 = k0 * cs - k1 * sn, kr1 = k0 * sn + k1 * cs;

    size_t hb = ((size_t)(b * H_ + h)) * 512 * 512;    // head base (elements)
    int lane2, e2;

    // Q (A-layout) and K (B-layout) share the same placement formula.
    int fs = s >> 4, rl = s & 15;
    int fragQK = fs * 4 + (d0 >> 5);
    int kl = d0 & 31;
    frag_place(rl, kl, lane2, e2);
    Qf[hb + (size_t)fragQK * 512 + lane2 * 16 + e2] = (__bf16)qr0;
    Kf[hb + (size_t)fragQK * 512 + lane2 * 16 + e2] = (__bf16)kr0;
    frag_place(rl, kl + 1, lane2, e2);
    Qf[hb + (size_t)fragQK * 512 + lane2 * 16 + e2] = (__bf16)qr1;
    Kf[hb + (size_t)fragQK * 512 + lane2 * 16 + e2] = (__bf16)kr1;

    // V (B-layout over key=32, dk=16 tiles)
    int fk2 = s >> 5, kls = s & 31;
    int fragV = fk2 * 8 + (d0 >> 4);
    frag_place(d0 & 15, kls, lane2, e2);
    Vf[hb + (size_t)fragV * 512 + lane2 * 16 + e2] = (__bf16)v0;
    frag_place((d0 + 1) & 15, kls, lane2, e2);
    Vf[hb + (size_t)fragV * 512 + lane2 * 16 + e2] = (__bf16)v1;
}

// ---------------------------------------------------------------------------
// Flash attention, causal + soft-mask reweighting.
// grid = (B*H, S/(16*8)), block = 256 (8 waves).  Each wave: 16 query rows.
// out = sum(e*m*v) / (sum(e*m) + 1e-6 * sum(e)),  e = exp(s - running_max).
// K/V fragment pointers advance by a constant 8 KiB per 32-key step, so all
// 16 fragment loads per step are base+immediate.
// ---------------------------------------------------------------------------
#define FA_WAVES 8
__global__ __launch_bounds__(256) void flash_attn_kernel(
    const __bf16* __restrict__ Qf, const __bf16* __restrict__ Kf,
    const __bf16* __restrict__ Vf, const float* __restrict__ soft_mask,
    __bf16* __restrict__ Ctxf)
{
    __shared__ __align__(16) __bf16 Plds[FA_WAVES][16][40];  // per-wave P tile

    const int lane = threadIdx.x & 31;
    const int wave = threadIdx.x >> 5;
    const int hf   = lane >> 4;
    const int nn   = lane & 15;
    const int bh   = blockIdx.x;
    const int b    = bh >> 4;
    const int h    = bh & 15;
    const int qt   = blockIdx.y * FA_WAVES + wave;   // query tile 0..127

    const __bf16* Qh = Qf + (size_t)bh * 512 * 512;
    const char*   pK = (const char*)(Kf + (size_t)bh * 512 * 512) + lane * 32;
    const char*   pV = (const char*)(Vf + (size_t)bh * 512 * 512) + lane * 32;

    bfx16 qfr[4];
#pragma unroll
    for (int fkd = 0; fkd < 4; ++fkd)
        qfr[fkd] = *(const bfx16*)((const char*)(Qh + ((size_t)qt * 4 + fkd) * 512) + lane * 32);

    fx8 oacc[8];
#pragma unroll
    for (int a = 0; a < 8; ++a) oacc[a] = fzero8();
    float m_run[8], l_pm[8], l_p[8];
#pragma unroll
    for (int r = 0; r < 8; ++r) { m_run[r] = -1e30f; l_pm[r] = 0.f; l_p[r] = 0.f; }

    const float scale = 0.08838834764831845f;        // 1/sqrt(128)
    const int kend = qt * 16 + 16;                   // causal limit (exclusive)

    for (int s0 = 0; s0 < kend; s0 += 32, pK += 8192, pV += 8192) {
        // ---- scores S = Q * K^T  (two 16x16 key tiles) -------------------
        fx8 sc[2];
#pragma unroll
        for (int t = 0; t < 2; ++t) {
            fx8 c = fzero8();
#pragma unroll
            for (int fkd = 0; fkd < 4; ++fkd) {
                bfx16 kfrag = *(const bfx16*)(pK + (t * 4 + fkd) * 1024);
                c = wmma_bf16(qfr[fkd], kfrag, c);
            }
            sc[t] = c;
        }

        // ---- scale + causal mask, block row max --------------------------
        float sval[2][8];
#pragma unroll
        for (int t = 0; t < 2; ++t)
#pragma unroll
            for (int r = 0; r < 8; ++r) {
                int qg = qt * 16 + r + 8 * hf;
                int kg = s0 + t * 16 + nn;
                sval[t][r] = (kg <= qg) ? sc[t][r] * scale : -1e30f;
            }

        float corr[8];
#pragma unroll
        for (int r = 0; r < 8; ++r) {
            float v = fmaxf(sval[0][r], sval[1][r]);
            v = fmaxf(v, __shfl_xor(v, 1, 32));
            v = fmaxf(v, __shfl_xor(v, 2, 32));
            v = fmaxf(v, __shfl_xor(v, 4, 32));
            v = fmaxf(v, __shfl_xor(v, 8, 32));
            float nm = fmaxf(m_run[r], v);
            corr[r] = __expf(m_run[r] - nm);
            m_run[r] = nm;
        }

        // ---- p = e^(s-m), pm = p * soft_mask; stage pm -> LDS ------------
        float rowp[8], rowpm[8];
#pragma unroll
        for (int r = 0; r < 8; ++r) { rowp[r] = 0.f; rowpm[r] = 0.f; }
#pragma unroll
        for (int t = 0; t < 2; ++t)
#pragma unroll
            for (int r = 0; r < 8; ++r) {
                int qg = qt * 16 + r + 8 * hf;
                int kg = s0 + t * 16 + nn;
                float pm = 0.f;
                if (kg <= qg) {
                    float p = __expf(sval[t][r] - m_run[r]);
                    pm = p * soft_mask[(size_t)qg * S_ + kg];
                    rowp[r] += p;
                    rowpm[r] += pm;
                }
                Plds[wave][r + 8 * hf][t * 16 + nn] = (__bf16)pm;
            }

#pragma unroll
        for (int r = 0; r < 8; ++r) {
            float sp = rowp[r], spm = rowpm[r];
            sp += __shfl_xor(sp, 1, 32);  spm += __shfl_xor(spm, 1, 32);
            sp += __shfl_xor(sp, 2, 32);  spm += __shfl_xor(spm, 2, 32);
            sp += __shfl_xor(sp, 4, 32);  spm += __shfl_xor(spm, 4, 32);
            sp += __shfl_xor(sp, 8, 32);  spm += __shfl_xor(spm, 8, 32);
            l_p[r]  = l_p[r]  * corr[r] + sp;
            l_pm[r] = l_pm[r] * corr[r] + spm;
        }
#pragma unroll
        for (int a = 0; a < 8; ++a)
#pragma unroll
            for (int r = 0; r < 8; ++r) oacc[a][r] *= corr[r];

        // same-wave LDS transpose round-trip: wait for our ds stores
        asm volatile("s_wait_dscnt 0x0" ::: "memory");

        // ---- read P as an A-fragment (lane = row, e -> key) --------------
        union { bfx16 v; bfx8 hh[2]; } pf;
        pf.hh[0] = *(const bfx8*)&Plds[wave][nn][hf * 8];
        pf.hh[1] = *(const bfx8*)&Plds[wave][nn][16 + hf * 8];

        // ---- O += P * V --------------------------------------------------
#pragma unroll
        for (int a = 0; a < 8; ++a) {
            bfx16 vfrag = *(const bfx16*)(pV + a * 1024);
            oacc[a] = wmma_bf16(pf.v, vfrag, oacc[a]);
        }
    }

    // ---- normalize and scatter into GEMM-A fragment layout of context ----
    float inv[8];
#pragma unroll
    for (int r = 0; r < 8; ++r)
        inv[r] = 1.f / (l_pm[r] + 1e-6f * l_p[r] + 1e-30f);

#pragma unroll
    for (int a = 0; a < 8; ++a)
#pragma unroll
        for (int r = 0; r < 8; ++r) {
            int m  = r + 8 * hf;                   // local query row
            int fm = b * 128 + qt;                 // global row tile
            int fk = h * 4 + (a >> 1);             // global k tile (32-wide)
            int kl = ((a & 1) << 4) + nn;          // k within tile
            int lane2, e2;
            frag_place(m, kl, lane2, e2);
            Ctxf[((size_t)fm * 64 + fk) * 512 + lane2 * 16 + e2] =
                (__bf16)(oacc[a][r] * inv[r]);
        }
}

// ---------------------------------------------------------------------------
// Host orchestration
// ---------------------------------------------------------------------------
extern "C" void kernel_launch(void* const* d_in, const int* in_sizes, int n_in,
                              void* d_out, int out_size, void* d_ws, size_t ws_size,
                              hipStream_t stream)
{
    (void)in_sizes; (void)n_in; (void)out_size; (void)ws_size;
    const float* x         = (const float*)d_in[0];
    const float* freqs     = (const float*)d_in[1];
    /* hard_mask (d_in[2]) unused: causal mask computed analytically */
    const float* soft_mask = (const float*)d_in[3];
    const float* wq = (const float*)d_in[4];  const float* bq = (const float*)d_in[5];
    const float* wk = (const float*)d_in[6];  const float* bk = (const float*)d_in[7];
    const float* wv = (const float*)d_in[8];  const float* bv = (const float*)d_in[9];
    const float* wo = (const float*)d_in[10]; const float* bo = (const float*)d_in[11];

    char* ws = (char*)d_ws;
    size_t off = 0;
    auto wsAlloc = [&](size_t bytes) -> void* {
        void* p = ws + off;
        off += (bytes + 255) & ~(size_t)255;
        return p;
    };
    const size_t MD = (size_t)MM_ * D_;   // 8.4M elements
    const size_t DD = (size_t)D_ * D_;

    __bf16* Xf   = (__bf16*)wsAlloc(MD * 2);
    __bf16* Wqf  = (__bf16*)wsAlloc(DD * 2);
    __bf16* Wkf  = (__bf16*)wsAlloc(DD * 2);
    __bf16* Wvf  = (__bf16*)wsAlloc(DD * 2);
    __bf16* Wof  = (__bf16*)wsAlloc(DD * 2);
    __bf16* Qp   = (__bf16*)wsAlloc(MD * 2);
    __bf16* Kp   = (__bf16*)wsAlloc(MD * 2);
    __bf16* Vp   = (__bf16*)wsAlloc(MD * 2);
    __bf16* Qf   = (__bf16*)wsAlloc(MD * 2);
    __bf16* Kf   = (__bf16*)wsAlloc(MD * 2);
    __bf16* Vf   = (__bf16*)wsAlloc(MD * 2);
    __bf16* Ctxf = (__bf16*)wsAlloc(MD * 2);

    // 1. swizzle x and weights into WMMA fragment layouts
    prep_swizzle_kernel<<<dim3((unsigned)(MD / 256)), 256, 0, stream>>>(x,  Xf,  MM_, D_);
    prep_swizzle_kernel<<<dim3((unsigned)(DD / 256)), 256, 0, stream>>>(wq, Wqf, D_,  D_);
    prep_swizzle_kernel<<<dim3((unsigned)(DD / 256)), 256, 0, stream>>>(wk, Wkf, D_,  D_);
    prep_swizzle_kernel<<<dim3((unsigned)(DD / 256)), 256, 0, stream>>>(wv, Wvf, D_,  D_);
    prep_swizzle_kernel<<<dim3((unsigned)(DD / 256)), 256, 0, stream>>>(wo, Wof, D_,  D_);

    // 2. QKV projections (WMMA GEMMs), block tile 256x128
    dim3 ggrid((MM_ / 256) * (D_ / 128));   // 256 blocks
    gemm_bf16_kernel<<<ggrid, 256, 0, stream>>>(Xf, Wqf, bq, Qp, 0);
    gemm_bf16_kernel<<<ggrid, 256, 0, stream>>>(Xf, Wkf, bk, Kp, 0);
    gemm_bf16_kernel<<<ggrid, 256, 0, stream>>>(Xf, Wvf, bv, Vp, 0);

    // 3. RoPE + scatter to per-head attention fragment layouts
    int ropeN = B_ * H_ * S_ * (DK_ / 2);   // 4.2M threads
    rope_scatter_kernel<<<dim3(ropeN / 256), 256, 0, stream>>>(
        Qp, Kp, Vp, freqs, Qf, Kf, Vf);

    // 4. flash attention (causal + soft-mask renorm), context in GEMM-A layout
    flash_attn_kernel<<<dim3(B_ * H_, S_ / (16 * FA_WAVES)), 256, 0, stream>>>(
        Qf, Kf, Vf, soft_mask, Ctxf);

    // 5. output projection -> f32 d_out
    gemm_bf16_kernel<<<ggrid, 256, 0, stream>>>(Ctxf, Wof, bo, d_out, 1);
}
